// NetworkAction_89713276879246
// MI455X (gfx1250) — matrix operand order repaired and checked by reading, not compiled, not run on hardware
//
#include <hip/hip_runtime.h>
#include <hip/hip_bf16.h>
#include <cstdint>

typedef __attribute__((ext_vector_type(16))) _Float16 v16h;
typedef __attribute__((ext_vector_type(8)))  float    v8f;

union V16 { v16h v; uint4 q[2]; };

#define INFF 3.402823466e+38f

// ---- workspace layout (bytes) ----
#define WS_IDX   0u                 // int [8192*32]            -> 1 MB
#define WS_HMAX  (1u<<20)           // _Float16 [8192*128]      -> 2 MB
#define WS_W     (3u<<20)           // packed f16 weights
// half-element offsets inside the W region (all 16B aligned)
#define OFF_BT2  0        // [128][64]   cw2^T
#define OFF_GT1  8192     // [64][160]   gw1^T, K padded 132->160 with zeros
#define OFF_GT2  18432    // [128][64]   gw2^T
#define OFF_GT3  26624    // [64][128]   gw3^T
#define OFF_GT4  34816    // [16][64]    gw4^T, N padded 4->16 with zeros

static constexpr int ACT_STR = 168;   // padded row stride (halfs): 336B, 16B aligned, bank-conflict free
static constexpr int H1_STR  = 72;    // 144B row stride for crowd layer-1 activations

__device__ __forceinline__ v8f wmma_f16(v16h a, v16h b, v8f c) {
  // D = A(16x32 f16) * B(32x16 f16) + C(16x16 f32)
  return __builtin_amdgcn_wmma_f32_16x16x32_f16(false, a, false, b, (short)0, c, false, false);
}

// ---------------------------------------------------------------------------
// Kernel 0: pack all weights into f16, transposed [n][k], padded for WMMA B tiles
// ---------------------------------------------------------------------------
__global__ __launch_bounds__(256) void pack_kernel(
    const float* __restrict__ cw2, const float* __restrict__ gw1,
    const float* __restrict__ gw2, const float* __restrict__ gw3,
    const float* __restrict__ gw4, _Float16* __restrict__ W) {
  const int t = threadIdx.x;
  for (int x = t; x < 128 * 64; x += 256) {            // BT2[n][k] = cw2[k][n]
    int n = x >> 6, k = x & 63;
    W[OFF_BT2 + x] = (_Float16)cw2[k * 128 + n];
  }
  for (int x = t; x < 64 * 160; x += 256) {            // GT1[n][k], k<132 valid
    int n = x / 160, k = x % 160;
    W[OFF_GT1 + x] = (k < 132) ? (_Float16)gw1[k * 64 + n] : (_Float16)0.0f;
  }
  for (int x = t; x < 128 * 64; x += 256) {            // GT2[n][k] = gw2[k][n]
    int n = x >> 6, k = x & 63;
    W[OFF_GT2 + x] = (_Float16)gw2[k * 128 + n];
  }
  for (int x = t; x < 64 * 128; x += 256) {            // GT3[n][k] = gw3[k][n]
    int n = x >> 7, k = x & 127;
    W[OFF_GT3 + x] = (_Float16)gw3[k * 64 + n];
  }
  for (int x = t; x < 16 * 64; x += 256) {             // GT4[n][k], n<4 valid
    int n = x >> 6, k = x & 63;
    W[OFF_GT4 + x] = (n < 4) ? (_Float16)gw4[k * 4 + n] : (_Float16)0.0f;
  }
}

// ---------------------------------------------------------------------------
// Kernel 1: top-K=32 nearest neighbors (squared 2D distance), one block/agent
// ---------------------------------------------------------------------------
__global__ __launch_bounds__(256) void topk_kernel(const float* __restrict__ s,
                                                   int* __restrict__ idxOut) {
  __shared__ float red_v[8];
  __shared__ int   red_j[8];
  __shared__ int   bro_j;
  const int tid  = threadIdx.x;
  const int lane = tid & 31;
  const int wv   = tid >> 5;
  const int i    = blockIdx.x;
  const float xi = s[i * 4 + 0], yi = s[i * 4 + 1];

  float v[32];
  #pragma unroll
  for (int m = 0; m < 32; ++m) {
    const int j = m * 256 + tid;
    const float dx = xi - s[j * 4 + 0];
    const float dy = yi - s[j * 4 + 1];
    v[m] = dx * dx + dy * dy;
  }
  unsigned taken = 0u;
  float lv = INFF; int lj = 0x7fffffff; int lm = 0;
  #pragma unroll
  for (int m = 0; m < 32; ++m)
    if (v[m] < lv) { lv = v[m]; lj = m * 256 + tid; lm = m; }

  for (int r = 0; r < 32; ++r) {
    float cv = lv; int cj = lj;
    #pragma unroll
    for (int off = 16; off > 0; off >>= 1) {
      float ov = __shfl_xor(cv, off, 32);
      int   oj = __shfl_xor(cj, off, 32);
      if (ov < cv || (ov == cv && oj < cj)) { cv = ov; cj = oj; }
    }
    if (lane == 0) { red_v[wv] = cv; red_j[wv] = cj; }
    __syncthreads();
    if (wv == 0) {
      float fv = (lane < 8) ? red_v[lane] : INFF;
      int   fj = (lane < 8) ? red_j[lane] : 0x7fffffff;
      #pragma unroll
      for (int off = 4; off > 0; off >>= 1) {
        float ov = __shfl_xor(fv, off, 32);
        int   oj = __shfl_xor(fj, off, 32);
        if (ov < fv || (ov == fv && oj < fj)) { fv = ov; fj = oj; }
      }
      if (lane == 0) { bro_j = fj; idxOut[i * 32 + r] = fj; }
    }
    __syncthreads();
    const int wj = bro_j;
    if (lj == wj) {               // exactly one thread owns the winner
      taken |= (1u << lm);
      lv = INFF; lj = 0x7fffffff;
      #pragma unroll
      for (int m = 0; m < 32; ++m)
        if (!((taken >> m) & 1u) && v[m] < lv) { lv = v[m]; lj = m * 256 + tid; lm = m; }
    }
  }
}

// ---------------------------------------------------------------------------
// Kernel 2: crowd MLP (5->64 VALU, 64->128 WMMA) + masked max-pool over K
// one wave per agent, 8 agents per block
// ---------------------------------------------------------------------------
__global__ __launch_bounds__(256) void crowd_kernel(
    const float* __restrict__ s, const int* __restrict__ idx,
    const float* __restrict__ cw1, const float* __restrict__ cb1,
    const float* __restrict__ cb2, const _Float16* __restrict__ BT2,
    _Float16* __restrict__ hmax) {
  __shared__ __align__(16) _Float16 h1[256 * H1_STR];
  const int lid = threadIdx.x & 31;
  const int wv  = threadIdx.x >> 5;
  const int i   = blockIdx.x * 8 + wv;          // agent, k = lid

  // ---- layer 1: x(5) -> h1(64), fp32 VALU ----
  const int j = idx[i * 32 + lid];
  const float x0 = s[i * 4 + 0] - s[j * 4 + 0];
  const float x1 = s[i * 4 + 1] - s[j * 4 + 1];
  const float x2 = s[i * 4 + 2] - s[j * 4 + 2];
  const float x3 = s[i * 4 + 3] - s[j * 4 + 3];
  const float tag = (j == i) ? 1.0f : 0.0f;
  const bool maskb = (x0 * x0 + x1 * x1) < 1.0f;
  const unsigned maskbits = (unsigned)__ballot(maskb);   // bit k = mask of row k

  _Float16* hrow = h1 + (wv * 32 + lid) * H1_STR;
  #pragma unroll
  for (int n = 0; n < 64; ++n) {
    float acc = cb1[n] + x0 * cw1[n] + x1 * cw1[64 + n] + x2 * cw1[128 + n] +
                x3 * cw1[192 + n] + tag * cw1[256 + n];
    hrow[n] = (_Float16)fmaxf(acc, 0.0f);
  }
  __syncthreads();

  // ---- layer 2: (32 rows x 64) @ (64 x 128) via WMMA f16, then mask+maxpool ----
  const int ln = lid & 15;
  const int hi = lid >> 4;
  V16 A[2][2];
  #pragma unroll
  for (int t = 0; t < 2; ++t) {
    #pragma unroll
    for (int ks = 0; ks < 2; ++ks) {
      const _Float16* p = h1 + (wv * 32 + t * 16 + ln) * H1_STR + ks * 32 + hi * 8;
      A[t][ks].q[0] = *(const uint4*)p;
      A[t][ks].q[1] = *(const uint4*)(p + 16);
    }
  }
  #pragma unroll
  for (int ct = 0; ct < 8; ++ct) {
    v8f c0 = {}, c1 = {};
    #pragma unroll
    for (int ks = 0; ks < 2; ++ks) {
      V16 B;
      const _Float16* bp = BT2 + (ct * 16 + ln) * 64 + ks * 32 + hi * 8;
      B.q[0] = *(const uint4*)bp;
      B.q[1] = *(const uint4*)(bp + 16);
      c0 = wmma_f16(A[0][ks].v, B.v, c0);
      c1 = wmma_f16(A[1][ks].v, B.v, c1);
    }
    const float b = cb2[ct * 16 + ln];
    float m = 0.0f;
    #pragma unroll
    for (int r = 0; r < 8; ++r) {
      const int k0 = r + hi * 8;                 // row-in-tile (== neighbor slot)
      float h0 = fmaxf(c0[r] + b, 0.0f);
      if (!((maskbits >> k0) & 1u)) h0 = 0.0f;
      float h2 = fmaxf(c1[r] + b, 0.0f);
      if (!((maskbits >> (16 + k0)) & 1u)) h2 = 0.0f;
      m = fmaxf(m, fmaxf(h0, h2));
    }
    m = fmaxf(m, __shfl_xor(m, 16, 32));         // merge the two lane halves
    if (lid < 16) hmax[i * 128 + ct * 16 + lid] = (_Float16)m;
  }
}

// ---------------------------------------------------------------------------
// Kernel 3: gain MLP, 4 WMMA layers; one wave handles 16 agents (one row-tile)
// ---------------------------------------------------------------------------
template <int KS, int CT, bool RELU, bool BGUARD>
__device__ __forceinline__ void mlp_layer(_Float16* act, const _Float16* BT,
                                          const float* bias, int lid) {
  const int ln  = lid & 15;
  const int hi  = lid >> 4;
  V16 A[KS];
  #pragma unroll
  for (int ks = 0; ks < KS; ++ks) {
    const _Float16* p = act + ln * ACT_STR + ks * 32 + hi * 8;
    A[ks].q[0] = *(const uint4*)p;
    A[ks].q[1] = *(const uint4*)(p + 16);
  }
  #pragma unroll
  for (int ct = 0; ct < CT; ++ct) {
    v8f c = {};
    #pragma unroll
    for (int ks = 0; ks < KS; ++ks) {
      V16 B;
      const _Float16* bp = BT + (ct * 16 + ln) * (KS * 32) + ks * 32 + hi * 8;
      B.q[0] = *(const uint4*)bp;
      B.q[1] = *(const uint4*)(bp + 16);
      c = wmma_f16(A[ks].v, B.v, c);
    }
    const int col = ct * 16 + ln;
    const float b = BGUARD ? ((col < 4) ? bias[col] : 0.0f) : bias[col];
    #pragma unroll
    for (int r = 0; r < 8; ++r) {
      float vv = c[r] + b;
      if (RELU) vv = fmaxf(vv, 0.0f);
      act[(r + hi * 8) * ACT_STR + col] = (_Float16)vv;
    }
  }
}

__global__ __launch_bounds__(256) void gain_kernel(
    const float* __restrict__ s, const float* __restrict__ g,
    const _Float16* __restrict__ W, const _Float16* __restrict__ hmax,
    const float* __restrict__ gb1, const float* __restrict__ gb2,
    const float* __restrict__ gb3, const float* __restrict__ gb4,
    float* __restrict__ out) {
  __shared__ __align__(16) _Float16 act_all[8 * 16 * ACT_STR];
  const int lid = threadIdx.x & 31;
  const int wv  = threadIdx.x >> 5;
  _Float16* act = act_all + wv * 16 * ACT_STR;
  const int aBase = blockIdx.x * 128 + wv * 16;

  float sgx = 0.f, sgy = 0.f, vx = 0.f, vy = 0.f;
  if (lid < 16) {
    const int a = aBase + lid;
    const uint4* src = (const uint4*)(hmax + a * 128);
    uint4* dst = (uint4*)(act + lid * ACT_STR);
    #pragma unroll
    for (int c = 0; c < 16; ++c) dst[c] = src[c];      // f[0..127] = h
    sgx = s[a * 4 + 0] - g[a * 2 + 0];
    sgy = s[a * 4 + 1] - g[a * 2 + 1];
    vx  = s[a * 4 + 2];
    vy  = s[a * 4 + 3];
    act[lid * ACT_STR + 128] = (_Float16)sgx;
    act[lid * ACT_STR + 129] = (_Float16)sgy;
    act[lid * ACT_STR + 130] = (_Float16)vx;
    act[lid * ACT_STR + 131] = (_Float16)vy;
    #pragma unroll
    for (int c = 132; c < 160; ++c) act[lid * ACT_STR + c] = (_Float16)0.0f;
  }
  // waves are independent; same-wave LDS ops stay in-order (DScnt)
  mlp_layer<5, 4, true,  false>(act, W + OFF_GT1, gb1, lid);  // 160 -> 64
  mlp_layer<2, 8, true,  false>(act, W + OFF_GT2, gb2, lid);  // 64  -> 128
  mlp_layer<4, 4, true,  false>(act, W + OFF_GT3, gb3, lid);  // 128 -> 64
  mlp_layer<2, 1, false, true >(act, W + OFF_GT4, gb4, lid);  // 64  -> 4 (padded 16)

  if (lid < 16) {
    const int a = aBase + lid;
    const float f0 = (float)act[lid * ACT_STR + 0];
    const float f1 = (float)act[lid * ACT_STR + 1];
    const float f2 = (float)act[lid * ACT_STR + 2];
    const float f3 = (float)act[lid * ACT_STR + 3];
    const float k0 = 2.0f / (1.0f + __expf(-f0)) + 0.2f;
    const float k1 = 2.0f / (1.0f + __expf(-f1)) + 0.2f;
    const float k2 = 2.0f / (1.0f + __expf(-f2)) + 0.2f;
    const float k3 = 2.0f / (1.0f + __expf(-f3)) + 0.2f;
    out[a * 2 + 0] = -k0 * sgx - k1 * vx;
    out[a * 2 + 1] = -k2 * sgy - k3 * vy;
  }
}

// ---------------------------------------------------------------------------
extern "C" void kernel_launch(void* const* d_in, const int* in_sizes, int n_in,
                              void* d_out, int out_size, void* d_ws, size_t ws_size,
                              hipStream_t stream) {
  const float* s   = (const float*)d_in[0];
  const float* g   = (const float*)d_in[1];
  const float* cw1 = (const float*)d_in[2];
  const float* cb1 = (const float*)d_in[3];
  const float* cw2 = (const float*)d_in[4];
  const float* cb2 = (const float*)d_in[5];
  const float* gw1 = (const float*)d_in[6];
  const float* gb1 = (const float*)d_in[7];
  const float* gw2 = (const float*)d_in[8];
  const float* gb2 = (const float*)d_in[9];
  const float* gw3 = (const float*)d_in[10];
  const float* gb3 = (const float*)d_in[11];
  const float* gw4 = (const float*)d_in[12];
  const float* gb4 = (const float*)d_in[13];

  char* ws = (char*)d_ws;
  int*      idx  = (int*)(ws + WS_IDX);
  _Float16* hmax = (_Float16*)(ws + WS_HMAX);
  _Float16* W    = (_Float16*)(ws + WS_W);
  float*    out  = (float*)d_out;

  pack_kernel<<<1, 256, 0, stream>>>(cw2, gw1, gw2, gw3, gw4, W);
  topk_kernel<<<8192, 256, 0, stream>>>(s, idx);
  crowd_kernel<<<1024, 256, 0, stream>>>(s, idx, cw1, cb1, cb2, W + OFF_BT2, hmax);
  gain_kernel<<<64, 256, 0, stream>>>(s, g, W, hmax, gb1, gb2, gb3, gb4, out);
}
